// RecurrentDynamics_42442866819494
// MI455X (gfx1250) — matrix operand order repaired and checked by reading, not compiled, not run on hardware
//
#include <hip/hip_runtime.h>
#include <math.h>

// RSSM: L=64 sequential steps. All matmuls via v_wmma_f32_16x16x32_bf16
// (f32 accumulate). Weights converted to bf16 once per launch (hot in 192MB
// L2); activations flow bf16 between GEMMs. GRU (both gate GEMMs + gate math)
// and distribution heads (GEMM + softplus/noise) are fused single kernels.
// Workgroups = 8 waves sharing one weight column-strip: all waves of a WG run
// on the same WGP, so W loads hit the shared WGP$ instead of re-streaming L2
// (~8x cut in L2 weight traffic; weights are re-read by every row block).

typedef __attribute__((ext_vector_type(16))) __bf16 v16bf;
typedef __attribute__((ext_vector_type(8)))  __bf16 v8bf;
typedef __attribute__((ext_vector_type(8)))  float  v8f;

#define WAVES 8   // waves per workgroup (row blocks sharing a column strip)

__device__ __forceinline__ v16bf afrag(const __bf16* __restrict__ p, int o0, int o1) {
    // A fragment (16x32): lane holds its row; e<8 -> K=o0+e, e>=8 -> K=o1+(e-8)
    v8bf lo = *(const v8bf*)(p + o0);
    v8bf hi = *(const v8bf*)(p + o1);
    return __builtin_shufflevector(lo, hi, 0, 1, 2, 3, 4, 5, 6, 7,
                                           8, 9, 10, 11, 12, 13, 14, 15);
}

#define WMMA_BF16(A_, B_, C_) \
    __builtin_amdgcn_wmma_f32_16x16x32_bf16(false, (A_), false, (B_), (short)0, (C_), false, false)

__device__ __forceinline__ float sigmoidf_(float x) { return 1.0f / (1.0f + __expf(-x)); }
__device__ __forceinline__ float softplusf_(float x) {
    return fmaxf(x, 0.0f) + log1pf(expf(-fabsf(x)));
}

// ---- generic GEMM + relu -> bf16 ------------------------------------------
// out[row,col] = relu( sum_k A[row,k]*W[col,k] + bias[col] ), bf16 out.
// 8 waves/WG; each wave does 16 rows x 64 cols; WG shares one 64-col W strip.
__global__ __launch_bounds__(32 * WAVES) void wmma_gemm_relu(
    const __bf16* __restrict__ A, int lda,
    const __bf16* __restrict__ W,
    const float* __restrict__ bias,
    __bf16* __restrict__ outB, int ldo,
    int K, int Nout)
{
    const int lane = threadIdx.x & 31;
    const int wave = threadIdx.x >> 5;
    const int half = lane >> 4, lr = lane & 15;
    const int rb   = blockIdx.y * WAVES + wave;
    const int row  = rb * 16 + lr;
    const int col0 = blockIdx.x * 64;

    const __bf16* arow = A + (size_t)row * lda;
    const __bf16* w0 = W + (size_t)(col0 + 0 * 16 + lr) * K;
    const __bf16* w1 = W + (size_t)(col0 + 1 * 16 + lr) * K;
    const __bf16* w2 = W + (size_t)(col0 + 2 * 16 + lr) * K;
    const __bf16* w3 = W + (size_t)(col0 + 3 * 16 + lr) * K;

    v8f acc0 = {}, acc1 = {}, acc2 = {}, acc3 = {};
    const int aoff0 = 16 * half, aoff1 = 16 * half + 8, boff = 16 * half;

#pragma unroll 2
    for (int kk = 0; kk < K; kk += 32) {
        v16bf a = afrag(arow, kk + aoff0, kk + aoff1);
        v16bf b0 = *(const v16bf*)(w0 + kk + boff);
        v16bf b1 = *(const v16bf*)(w1 + kk + boff);
        v16bf b2 = *(const v16bf*)(w2 + kk + boff);
        v16bf b3 = *(const v16bf*)(w3 + kk + boff);
        acc0 = WMMA_BF16(a, b0, acc0);
        acc1 = WMMA_BF16(a, b1, acc1);
        acc2 = WMMA_BF16(a, b2, acc2);
        acc3 = WMMA_BF16(a, b3, acc3);
    }

    const int orow0 = rb * 16 + 8 * half;   // D: M = r + 8*(lane>>4)
    const float bv0 = bias[col0 + 0 * 16 + lr];
    const float bv1 = bias[col0 + 1 * 16 + lr];
    const float bv2 = bias[col0 + 2 * 16 + lr];
    const float bv3 = bias[col0 + 3 * 16 + lr];
#pragma unroll
    for (int r = 0; r < 8; ++r) {
        const size_t o = (size_t)(orow0 + r) * ldo + col0 + lr;
        outB[o + 0]  = (__bf16)fmaxf(acc0[r] + bv0, 0.0f);
        outB[o + 16] = (__bf16)fmaxf(acc1[r] + bv1, 0.0f);
        outB[o + 32] = (__bf16)fmaxf(acc2[r] + bv2, 0.0f);
        outB[o + 48] = (__bf16)fmaxf(acc3[r] + bv3, 0.0f);
    }
}

// ---- fused GRU: gi = x@Wih^T+bih, gh = h@Whh^T+bhh, then gate math --------
// Wave computes a 16-row x 32-col strip of ALL THREE gates for BOTH matrices
// (12 accumulators), then h_new = (1-z)*tanh(i_n + r*h_n) + z*h in-register.
__global__ __launch_bounds__(32 * WAVES) void gru_fused(
    const __bf16* __restrict__ X,      // B x Hn (lda = Hn)
    const __bf16* __restrict__ Hp,     // B x HQ, cols [0,Hn) = h_prev bf16
    const float*  __restrict__ hprevf, // B x Hn f32
    const __bf16* __restrict__ Wih, const __bf16* __restrict__ Whh, // 3Hn x Hn
    const float* __restrict__ bih, const float* __restrict__ bhh,
    float* __restrict__ hout, __bf16* __restrict__ hq,
    int Hn, int HQ)
{
    const int lane = threadIdx.x & 31;
    const int wave = threadIdx.x >> 5;
    const int half = lane >> 4, lr = lane & 15;
    const int rb   = blockIdx.y * WAVES + wave;
    const int row  = rb * 16 + lr;
    const int c0   = blockIdx.x * 32;

    const __bf16* xrow = X  + (size_t)row * Hn;
    const __bf16* hrow = Hp + (size_t)row * HQ;

    const __bf16* wi[3][2];
    const __bf16* wh[3][2];
#pragma unroll
    for (int g = 0; g < 3; ++g)
#pragma unroll
        for (int j = 0; j < 2; ++j) {
            wi[g][j] = Wih + (size_t)(g * Hn + c0 + 16 * j + lr) * Hn;
            wh[g][j] = Whh + (size_t)(g * Hn + c0 + 16 * j + lr) * Hn;
        }

    v8f aI[3][2] = {}, aH[3][2] = {};
    const int aoff0 = 16 * half, aoff1 = 16 * half + 8, boff = 16 * half;

    for (int kk = 0; kk < Hn; kk += 32) {
        v16bf ax = afrag(xrow, kk + aoff0, kk + aoff1);
        v16bf ah = afrag(hrow, kk + aoff0, kk + aoff1);
#pragma unroll
        for (int g = 0; g < 3; ++g)
#pragma unroll
            for (int j = 0; j < 2; ++j) {
                v16bf bi = *(const v16bf*)(wi[g][j] + kk + boff);
                aI[g][j] = WMMA_BF16(ax, bi, aI[g][j]);
                v16bf bh = *(const v16bf*)(wh[g][j] + kk + boff);
                aH[g][j] = WMMA_BF16(ah, bh, aH[g][j]);
            }
    }

    const int orow0 = rb * 16 + 8 * half;
#pragma unroll
    for (int j = 0; j < 2; ++j) {
        const int col = c0 + 16 * j + lr;
        const float biR = bih[col], biZ = bih[Hn + col], biN = bih[2 * Hn + col];
        const float bhR = bhh[col], bhZ = bhh[Hn + col], bhN = bhh[2 * Hn + col];
#pragma unroll
        for (int r = 0; r < 8; ++r) {
            const int orow = orow0 + r;
            float rg = sigmoidf_((aI[0][j][r] + biR) + (aH[0][j][r] + bhR));
            float zg = sigmoidf_((aI[1][j][r] + biZ) + (aH[1][j][r] + bhZ));
            float ng = tanhf((aI[2][j][r] + biN) + rg * (aH[2][j][r] + bhN));
            float h  = hprevf[(size_t)orow * Hn + col];
            float v  = (1.0f - zg) * ng + zg * h;
            hout[(size_t)orow * Hn + col] = v;
            hq[(size_t)orow * HQ + col]   = (__bf16)v;
        }
    }
}

// ---- fused distribution head ----------------------------------------------
// tmp = A @ W^T + bias (Nout = 2S). Wave computes mean cols [c0,c0+32) and
// std cols [S+c0,S+c0+32) so the softplus/noise epilogue pairs them in-register.
__global__ __launch_bounds__(32 * WAVES) void head_fused(
    const __bf16* __restrict__ A, int lda,
    const __bf16* __restrict__ W,     // 2S x K
    const float* __restrict__ bias,   // 2S
    const float* __restrict__ noise,  // B x S
    float* __restrict__ om, float* __restrict__ os, float* __restrict__ ost,
    int K, int Sn)
{
    const int lane = threadIdx.x & 31;
    const int wave = threadIdx.x >> 5;
    const int half = lane >> 4, lr = lane & 15;
    const int rb   = blockIdx.y * WAVES + wave;
    const int row  = rb * 16 + lr;
    const int c0   = blockIdx.x * 32;

    const __bf16* arow = A + (size_t)row * lda;
    const __bf16* wm0 = W + (size_t)(c0 + 0 * 16 + lr) * K;
    const __bf16* wm1 = W + (size_t)(c0 + 1 * 16 + lr) * K;
    const __bf16* ws0 = W + (size_t)(Sn + c0 + 0 * 16 + lr) * K;
    const __bf16* ws1 = W + (size_t)(Sn + c0 + 1 * 16 + lr) * K;

    v8f am0 = {}, am1 = {}, as0 = {}, as1 = {};
    const int aoff0 = 16 * half, aoff1 = 16 * half + 8, boff = 16 * half;

#pragma unroll 2
    for (int kk = 0; kk < K; kk += 32) {
        v16bf a = afrag(arow, kk + aoff0, kk + aoff1);
        v16bf b0 = *(const v16bf*)(wm0 + kk + boff);
        v16bf b1 = *(const v16bf*)(wm1 + kk + boff);
        v16bf b2 = *(const v16bf*)(ws0 + kk + boff);
        v16bf b3 = *(const v16bf*)(ws1 + kk + boff);
        am0 = WMMA_BF16(a, b0, am0);
        am1 = WMMA_BF16(a, b1, am1);
        as0 = WMMA_BF16(a, b2, as0);
        as1 = WMMA_BF16(a, b3, as1);
    }

    const int orow0 = rb * 16 + 8 * half;
#pragma unroll
    for (int j = 0; j < 2; ++j) {
        const int col = c0 + 16 * j + lr;
        const float bm = bias[col], bsv = bias[Sn + col];
        const v8f& aM = j ? am1 : am0;
        const v8f& aS = j ? as1 : as0;
#pragma unroll
        for (int r = 0; r < 8; ++r) {
            const size_t o = (size_t)(orow0 + r) * Sn + col;
            float m  = aM[r] + bm;
            float sp = softplusf_(aS[r] + bsv) + 0.1f;
            om[o]  = m;
            os[o]  = sp;
            ost[o] = m + sp * noise[o];
        }
    }
}

// ---- small elementwise kernels -------------------------------------------

__global__ void cvt_bf16(const float* __restrict__ src, __bf16* __restrict__ dst, int n) {
    int i = blockIdx.x * blockDim.x + threadIdx.x;
    if (i < n) dst[i] = (__bf16)src[i];
}

__global__ void cvt_bf16_strided(const float* __restrict__ src, __bf16* __restrict__ dst,
                                 int rows, int cols, int ldd) {
    int i = blockIdx.x * blockDim.x + threadIdx.x;
    if (i >= rows * cols) return;
    int b = i / cols, c = i - b * cols;
    dst[(size_t)b * ldd + c] = (__bf16)src[i];
}

// xa = bf16([s_post * nt, actions])  (B x (S+A))
__global__ void build_xa(const float* __restrict__ spost, const float* __restrict__ nt,
                         const float* __restrict__ act, __bf16* __restrict__ xa,
                         int Bn, int Sn, int An) {
    int KA = Sn + An;
    int i = blockIdx.x * blockDim.x + threadIdx.x;
    if (i >= Bn * KA) return;
    int b = i / KA, c = i - b * KA;
    float v = (c < Sn) ? spost[(size_t)b * Sn + c] * nt[b]
                       : act[(size_t)b * An + (c - Sn)];
    xa[i] = (__bf16)v;
}

// ---- host ----------------------------------------------------------------

extern "C" void kernel_launch(void* const* d_in, const int* in_sizes, int n_in,
                              void* d_out, int out_size, void* d_ws, size_t ws_size,
                              hipStream_t stream) {
    constexpr int Ln = 64, Bn = 512, Hn = 1024, Sn = 128, An = 32, Nn = 1024, En = 1024;
    constexpr int KA = Sn + An;          // 160
    constexpr int HQ = Hn + En;          // 2048

    const float* prev_hidden = (const float*)d_in[0];
    const float* prev_state  = (const float*)d_in[1];
    const float* actions     = (const float*)d_in[2];
    const float* obs         = (const float*)d_in[3];
    const float* non_terms   = (const float*)d_in[4];
    const float* pr_noise    = (const float*)d_in[5];
    const float* po_noise    = (const float*)d_in[6];
    const float* W_sa = (const float*)d_in[7];  const float* b_sa = (const float*)d_in[8];
    const float* W_ih = (const float*)d_in[9];  const float* W_hh = (const float*)d_in[10];
    const float* b_ih = (const float*)d_in[11]; const float* b_hh = (const float*)d_in[12];
    const float* W_ep = (const float*)d_in[13]; const float* b_ep = (const float*)d_in[14];
    const float* W_pr = (const float*)d_in[15]; const float* b_pr = (const float*)d_in[16];
    const float* W_eq = (const float*)d_in[17]; const float* b_eq = (const float*)d_in[18];
    const float* W_po = (const float*)d_in[19]; const float* b_po = (const float*)d_in[20];

    float* out = (float*)d_out;
    const size_t stepH = (size_t)Bn * Hn;
    const size_t bs    = (size_t)Bn * Sn;
    float* o_hid = out;
    float* o_pm  = o_hid + (size_t)Ln * stepH;
    float* o_ps  = o_pm  + (size_t)Ln * bs;
    float* o_pst = o_ps  + (size_t)Ln * bs;
    float* o_qm  = o_pst + (size_t)Ln * bs;
    float* o_qs  = o_qm  + (size_t)Ln * bs;
    float* o_qst = o_qs  + (size_t)Ln * bs;

    // Workspace: bf16 weights + bf16 activation staging
    __bf16* wp = (__bf16*)d_ws;
    __bf16* wsa = wp; wp += (size_t)Hn * KA;
    __bf16* wih = wp; wp += (size_t)3 * Hn * Hn;
    __bf16* whh = wp; wp += (size_t)3 * Hn * Hn;
    __bf16* wep = wp; wp += (size_t)Nn * Hn;
    __bf16* wpr = wp; wp += (size_t)2 * Sn * Nn;
    __bf16* weq = wp; wp += (size_t)Nn * HQ;
    __bf16* wpo = wp; wp += (size_t)2 * Sn * Nn;
    __bf16* xa  = wp; wp += (size_t)Bn * KA;
    __bf16* xb  = wp; wp += (size_t)Bn * Hn;
    __bf16* hq0 = wp; wp += (size_t)Bn * HQ;
    __bf16* hq1 = wp; wp += (size_t)Bn * HQ;
    __bf16* peb = wp; wp += (size_t)Bn * Nn;
    __bf16* qeb = wp; wp += (size_t)Bn * Nn;

    const dim3 blk(32 * WAVES, 1, 1);               // 8 waves / WG
    const int  rby = Bn / (16 * WAVES);             // 4 row groups
    const dim3 grid_x (Hn / 64, rby);               // embed/pe/qe GEMMs
    const dim3 grid_gr(Hn / 32, rby);               // fused GRU (32-col strips)
    const dim3 grid_hd(Sn / 32, rby);               // fused heads
    const int ew = 256;
    auto nb = [](int n) { return (n + 255) / 256; };

    // per-launch weight conversion fp32 -> bf16 (stays hot in L2)
    cvt_bf16<<<nb(Hn * KA),     ew, 0, stream>>>(W_sa, wsa, Hn * KA);
    cvt_bf16<<<nb(3 * Hn * Hn), ew, 0, stream>>>(W_ih, wih, 3 * Hn * Hn);
    cvt_bf16<<<nb(3 * Hn * Hn), ew, 0, stream>>>(W_hh, whh, 3 * Hn * Hn);
    cvt_bf16<<<nb(Nn * Hn),     ew, 0, stream>>>(W_ep, wep, Nn * Hn);
    cvt_bf16<<<nb(2 * Sn * Nn), ew, 0, stream>>>(W_pr, wpr, 2 * Sn * Nn);
    cvt_bf16<<<nb(Nn * HQ),     ew, 0, stream>>>(W_eq, weq, Nn * HQ);
    cvt_bf16<<<nb(2 * Sn * Nn), ew, 0, stream>>>(W_po, wpo, 2 * Sn * Nn);
    cvt_bf16_strided<<<nb(Bn * Hn), ew, 0, stream>>>(prev_hidden, hq1, Bn, Hn, HQ);

    for (int t = 0; t < Ln; ++t) {
        __bf16* hq_cur  = (t & 1) ? hq1 : hq0;
        __bf16* hq_prev = (t & 1) ? hq0 : hq1;
        const float* st_src = (t == 0) ? prev_state  : (o_qst + (size_t)(t - 1) * bs);
        const float* hprevf = (t == 0) ? prev_hidden : (o_hid + (size_t)(t - 1) * stepH);
        const float* nt     = non_terms + (size_t)t * Bn;
        const float* act_t  = actions   + (size_t)t * Bn * An;
        const float* obs_t  = obs       + (size_t)t * Bn * En;
        float* hnew = o_hid + (size_t)t * stepH;

        build_xa<<<nb(Bn * KA), ew, 0, stream>>>(st_src, nt, act_t, xa, Bn, Sn, An);
        cvt_bf16_strided<<<nb(Bn * En), ew, 0, stream>>>(obs_t, hq_cur + Hn, Bn, En, HQ);

        // x = relu(xa @ W_sa^T + b_sa)
        wmma_gemm_relu<<<grid_x, blk, 0, stream>>>(xa, KA, wsa, b_sa, xb, Hn, KA, Hn);
        // fused GRU: both gate GEMMs + nonlinearity -> h_new (f32 + bf16)
        gru_fused<<<grid_gr, blk, 0, stream>>>(xb, hq_prev, hprevf, wih, whh,
                                               b_ih, b_hh, hnew, hq_cur, Hn, HQ);
        // prior
        wmma_gemm_relu<<<grid_x, blk, 0, stream>>>(hq_cur, HQ, wep, b_ep, peb, Nn, Hn, Nn);
        head_fused<<<grid_hd, blk, 0, stream>>>(peb, Nn, wpr, b_pr,
                                                pr_noise + (size_t)t * bs,
                                                o_pm + (size_t)t * bs, o_ps + (size_t)t * bs,
                                                o_pst + (size_t)t * bs, Nn, Sn);
        // posterior ([h, obs] via hq_cur, K = 2048)
        wmma_gemm_relu<<<grid_x, blk, 0, stream>>>(hq_cur, HQ, weq, b_eq, qeb, Nn, HQ, Nn);
        head_fused<<<grid_hd, blk, 0, stream>>>(qeb, Nn, wpo, b_po,
                                                po_noise + (size_t)t * bs,
                                                o_qm + (size_t)t * bs, o_qs + (size_t)t * bs,
                                                o_qst + (size_t)t * bs, Nn, Sn);
    }
    (void)in_sizes; (void)n_in; (void)out_size; (void)ws_size;
}
